// ASTGCN_34282428957250
// MI455X (gfx1250) — matrix operand order repaired and checked by reading, not compiled, not run on hardware
//
#include <hip/hip_runtime.h>
#include <hip/hip_bf16.h>

// ---------------- problem constants ----------------
constexpr int Bb  = 16;
constexpr int Nn  = 1024;
constexpr int Fin = 3;
constexpr int Tt  = 4;
constexpr int Cc  = 64;
constexpr int Ee  = 16384;
constexpr int DH  = Nn + 6 * Tt;   // 1048 head input

typedef __bf16 bf16;
typedef __attribute__((ext_vector_type(8)))  bf16  v8bf;
typedef __attribute__((ext_vector_type(16))) bf16  v16bf;
typedef __attribute__((ext_vector_type(8)))  float v8f;

#define GIDX (blockIdx.x * blockDim.x + threadIdx.x)

// ---------------- pooling ----------------
__global__ void scores_kernel(const float* __restrict__ x, const float* __restrict__ pw,
                              float* __restrict__ scores) {
    int idx = GIDX; if (idx >= Bb * Nn * Tt) return;
    int t = idx % Tt; int n = (idx / Tt) % Nn; int b = idx / (Tt * Nn);
    float nrm = sqrtf(pw[0]*pw[0] + pw[1]*pw[1] + pw[2]*pw[2]);
    float s = 0.f;
    for (int f = 0; f < Fin; ++f) s += x[((b*Nn + n)*Fin + f)*Tt + t] * pw[f];
    scores[idx] = tanhf(s / nrm);
}

// bitonic descending sort of 1024 scores per (b,t); block = 1024 threads
__global__ void pool_sort_kernel(const float* __restrict__ scores,
                                 float* __restrict__ sp, int* __restrict__ perm) {
    __shared__ float sv[Nn];
    __shared__ int   si[Nn];
    int b = blockIdx.x / Tt, t = blockIdx.x % Tt;
    int tid = threadIdx.x;
    sv[tid] = scores[(b*Nn + tid)*Tt + t];
    si[tid] = tid;
    __syncthreads();
    for (int k = 2; k <= Nn; k <<= 1) {
        for (int j = k >> 1; j > 0; j >>= 1) {
            int ixj = tid ^ j;
            if (ixj > tid) {
                bool desc = ((tid & k) == 0);
                float a = sv[tid], c = sv[ixj];
                bool doSwap = desc ? (a < c) : (a > c);
                if (doSwap) {
                    sv[tid] = c; sv[ixj] = a;
                    int ti = si[tid]; si[tid] = si[ixj]; si[ixj] = ti;
                }
            }
            __syncthreads();
        }
    }
    sp[(b*Nn + tid)*Tt + t]   = sv[tid];
    perm[(b*Nn + tid)*Tt + t] = si[tid];
}

__global__ void gather_kernel(const float* __restrict__ x, const int* __restrict__ perm,
                              const float* __restrict__ sp, float* __restrict__ h0) {
    int idx = GIDX; if (idx >= Bb * Nn * Fin * Tt) return;
    int t = idx % Tt; int f = (idx / Tt) % Fin; int n = (idx / (Tt*Fin)) % Nn;
    int b = idx / (Tt*Fin*Nn);
    int src = perm[(b*Nn + n)*Tt + t];
    h0[idx] = x[((b*Nn + src)*Fin + f)*Tt + t] * sp[(b*Nn + n)*Tt + t];
}

__global__ void inv_kernel(const int* __restrict__ perm, int* __restrict__ invp) {
    int idx = GIDX; if (idx >= Tt * Nn) return;
    int t = idx / Nn, n2 = idx % Nn;
    int p = perm[(t*Nn + n2)*Tt + (Tt - 1)];   // batch index == t quirk of reference
    invp[t*Nn + p] = n2;
}

__global__ void edges_map_kernel(const int* __restrict__ edge_index, const int* __restrict__ invp,
                                 int* __restrict__ edges_t) {
    int idx = GIDX; if (idx >= Tt * 2 * Ee) return;
    int t = idx / (2*Ee); int r = idx % (2*Ee); int d = r / Ee; int e = r % Ee;
    edges_t[idx] = invp[t*Nn + edge_index[d*Ee + e]];
}

__global__ void zero_kernel(float* p, long long n) {
    long long i = (long long)blockIdx.x * blockDim.x + threadIdx.x;
    if (i < n) p[i] = 0.f;
}

__global__ void deg_kernel(const int* __restrict__ edges_t, float* deg) {
    int idx = GIDX; if (idx >= Tt * Ee) return;
    int t = idx / Ee, e = idx % Ee;
    atomicAdd(&deg[t*Nn + edges_t[(t*2 + 0)*Ee + e]], 1.0f);
}

__global__ void dinv_kernel(float* deg) {
    int idx = GIDX; if (idx >= Tt * Nn) return;
    float d = deg[idx];
    deg[idx] = (d > 0.f) ? rsqrtf(d) : 0.f;
}

// ---------------- temporal attention (tiny) ----------------
__global__ void t_lhs1_kernel(const float* __restrict__ X, const float* __restrict__ U1,
                              float* __restrict__ lhs_tf, int f) {
    int idx = GIDX; if (idx >= Bb * Tt * f) return;
    int c = idx % f; int t = (idx / f) % Tt; int b = idx / (f*Tt);
    float s = 0.f;
    for (int n = 0; n < Nn; ++n) s += X[((b*Nn + n)*f + c)*Tt + t] * U1[n];
    lhs_tf[idx] = s;
}

__global__ void t_lhs2_kernel(const float* __restrict__ lhs_tf, const float* __restrict__ U2,
                              float* __restrict__ lhsN, int f) {
    int idx = GIDX; if (idx >= Bb * Tt * Nn) return;
    int n = idx % Nn; int t = (idx / Nn) % Tt; int b = idx / (Nn*Tt);
    float s = 0.f;
    for (int c = 0; c < f; ++c) s += lhs_tf[(b*Tt + t)*f + c] * U2[c*Nn + n];
    lhsN[idx] = s;
}

__global__ void t_rhs_kernel(const float* __restrict__ X, const float* __restrict__ U3,
                             float* __restrict__ rhsNT, int f) {
    int idx = GIDX; if (idx >= Bb * Nn * Tt) return;
    int t = idx % Tt; int n = (idx / Tt) % Nn; int b = idx / (Tt*Nn);
    float s = 0.f;
    for (int c = 0; c < f; ++c) s += U3[c] * X[((b*Nn + n)*f + c)*Tt + t];
    rhsNT[idx] = s;
}

__global__ void et_kernel(const float* __restrict__ lhsN, const float* __restrict__ rhsNT,
                          const float* __restrict__ Ve, const float* __restrict__ be,
                          float* __restrict__ Et) {
    __shared__ float prod[Tt][Tt];
    __shared__ float e1[Tt][Tt];
    int b = blockIdx.x;
    int tid = threadIdx.x;
    if (tid < Tt*Tt) {
        int i = tid / Tt, j = tid % Tt;
        float s = 0.f;
        for (int n = 0; n < Nn; ++n) s += lhsN[(b*Tt + i)*Nn + n] * rhsNT[(b*Nn + n)*Tt + j];
        prod[i][j] = 1.f / (1.f + __expf(-(s + be[i*Tt + j])));
    }
    __syncthreads();
    if (tid < Tt*Tt) {
        int i = tid / Tt, j = tid % Tt;
        float s = 0.f;
        for (int k = 0; k < Tt; ++k) s += Ve[i*Tt + k] * prod[k][j];
        e1[i][j] = s;
    }
    __syncthreads();
    if (tid < Tt) {           // softmax over axis=1 (rows i) per column j
        int j = tid;
        float m = -3.4e38f;
        for (int i = 0; i < Tt; ++i) m = fmaxf(m, e1[i][j]);
        float ex[Tt]; float sum = 0.f;
        for (int i = 0; i < Tt; ++i) { ex[i] = __expf(e1[i][j] - m); sum += ex[i]; }
        for (int i = 0; i < Tt; ++i) Et[(b*Tt + i)*Tt + j] = ex[i] / sum;
    }
}

__global__ void xtat_kernel(const float* __restrict__ X, const float* __restrict__ Et,
                            float* __restrict__ xtat, int f) {
    int idx = GIDX; if (idx >= Bb * Nn * f * Tt) return;
    int t = idx % Tt; int c = (idx / Tt) % f; int n = (idx / (Tt*f)) % Nn; int b = idx / (Tt*f*Nn);
    float s = 0.f;
    for (int tt = 0; tt < Tt; ++tt)
        s += X[((b*Nn + n)*f + c)*Tt + tt] * Et[(b*Tt + tt)*Tt + t];
    xtat[idx] = s;
}

// ---------------- spatial attention ----------------
__global__ void lhs_s_kernel(const float* __restrict__ xtat, const float* __restrict__ W1,
                             const float* __restrict__ W2, float* __restrict__ lhs_s, int f) {
    int idx = GIDX; if (idx >= Bb * Nn * Tt) return;
    int j = idx % Tt; int n = (idx / Tt) % Nn; int b = idx / (Tt*Nn);
    float s = 0.f;
    for (int c = 0; c < f; ++c) {
        float tmp = 0.f;
        for (int t = 0; t < Tt; ++t) tmp += xtat[((b*Nn + n)*f + c)*Tt + t] * W1[t];
        s += tmp * W2[c*Tt + j];
    }
    lhs_s[(b*Nn + n)*Tt + j] = s;
}

__global__ void rhs_s_kernel(const float* __restrict__ xtat, const float* __restrict__ W3,
                             float* __restrict__ rhs_s, int f) {
    int idx = GIDX; if (idx >= Bb * Tt * Nn) return;
    int n = idx % Nn; int t = (idx / Nn) % Tt; int b = idx / (Nn*Tt);
    float s = 0.f;
    for (int c = 0; c < f; ++c) s += W3[c] * xtat[((b*Nn + n)*f + c)*Tt + t];
    rhs_s[(b*Tt + t)*Nn + n] = s;
}

// P[i][j] = sigmoid(sum_t lhs_s[b,i,t]*rhs_s[b,t,j] + bs[i][j]); stored transposed in bf16
__global__ void pt_kernel(const float* __restrict__ lhs_s, const float* __restrict__ rhs_s,
                          const float* __restrict__ bs, bf16* __restrict__ Pt) {
    long long idx = (long long)blockIdx.x * blockDim.x + threadIdx.x;
    if (idx >= (long long)Bb * Nn * Nn) return;
    int i = (int)(idx % Nn); int j = (int)((idx / Nn) % Nn); int b = (int)(idx / ((long long)Nn*Nn));
    float s = 0.f;
    for (int t = 0; t < Tt; ++t)
        s += lhs_s[(b*Nn + i)*Tt + t] * rhs_s[(b*Tt + t)*Nn + j];
    float p = 1.f / (1.f + __expf(-(s + bs[(long long)i*Nn + j])));
    Pt[(long long)b*Nn*Nn + (long long)j*Nn + i] = (bf16)p;
}

__global__ void f32_to_bf16_kernel(const float* __restrict__ src, bf16* __restrict__ dst, long long n) {
    long long i = (long long)blockIdx.x * blockDim.x + threadIdx.x;
    if (i < n) dst[i] = (bf16)src[i];
}

// S[b] = Vs(bf16) @ P[b].
// Workgroup = 8 waves covering 128 rows x 16 cols. The 16-column B panel
// (16 x 1024 bf16 = 32 KB, contiguous in transposed Pt) is shared by all 8
// waves, so it is staged ONCE into LDS via global_load_async_to_lds_b128
// (ASYNCcnt), then each wave reads B fragments from LDS (ds_load_b128) while
// A (Vs rows, unique per wave) streams from global with prefetch.
__global__ void __launch_bounds__(256)
wmma_vs_p_kernel(const bf16* __restrict__ A,   // Vs bf16, (N,N) row-major
                 const bf16* __restrict__ Pt,  // (B,N,N): Pt[b][col][k] = P[b][k][col]
                 float* __restrict__ S) {
    __shared__ bf16 ldsB[16 * Nn];             // 32 KB B panel: [col_local][k]
    const int tid   = threadIdx.x;
    const int lane  = tid & 31;
    const int wave  = tid >> 5;
    const int bIdx  = blockIdx.z;
    const int mBase = (blockIdx.y * 8 + wave) * 16;
    const int nBase = blockIdx.x * 16;
    const int half  = lane >> 4;          // lane group 0/1
    const int l16   = lane & 15;

    // ---- async copy of the 32 KB B panel: global -> LDS ----
    {
        unsigned ldsBase = (unsigned)(size_t)&ldsB[0];
        const char* gBase = (const char*)(Pt + (size_t)bIdx * Nn * Nn + (size_t)nBase * Nn);
#pragma unroll
        for (int it = 0; it < 8; ++it) {
            int q = it * 256 + tid;                       // 2048 x 16B chunks
            unsigned ldsAddr = ldsBase + q * 16;
            unsigned long long ga = (unsigned long long)(size_t)(gBase + q * 16);
            asm volatile("global_load_async_to_lds_b128 %0, %1, off"
                         :: "v"(ldsAddr), "v"(ga) : "memory");
        }
        asm volatile("s_wait_asynccnt 0x0" ::: "memory");
    }
    __syncthreads();

    // ---- WMMA main loop ----
    const int kA = half * 8;              // A frag: lanes 0-15 -> K 0..7/16..23, 16-31 -> 8..15/24..31
    const int kB = half * 8;              // B frag: lanes 0-15 -> K 0..7/16..23, 16-31 -> 8..15/24..31
    const bf16* aPtr = A + (size_t)(mBase + l16) * Nn;
    const bf16* bLds = &ldsB[l16 * Nn];
    v8f acc = {};
    for (int k = 0; k < Nn; k += 32) {
        v8bf a0 = *(const v8bf*)(aPtr + k + kA);
        v8bf a1 = *(const v8bf*)(aPtr + k + kA + 16);
        v8bf b0 = *(const v8bf*)(bLds + k + kB);
        v8bf b1 = *(const v8bf*)(bLds + k + kB + 16);
        v16bf a, b;
#pragma unroll
        for (int i = 0; i < 8; ++i) { a[i] = a0[i]; a[i+8] = a1[i]; b[i] = b0[i]; b[i+8] = b1[i]; }
        acc = __builtin_amdgcn_wmma_f32_16x16x32_bf16(false, a, false, b, (short)0, acc, false, false);
        if (k + 32 < Nn) __builtin_prefetch(aPtr + k + 32 + kA, 0, 1);
    }
    float* dst = S + (size_t)bIdx * Nn * Nn + nBase + l16;
#pragma unroll
    for (int r = 0; r < 8; ++r) {
        int row = mBase + half * 8 + r;
        dst[(size_t)row * Nn] = acc[r];
    }
}

// column-wise softmax (axis=1 of (B,N,N)) in place
__global__ void colsoftmax_kernel(float* __restrict__ S) {
    __shared__ float red[128];
    int b = blockIdx.x / Nn, j = blockIdx.x % Nn;
    float* col = S + (size_t)b*Nn*Nn + j;
    int tid = threadIdx.x;
    float m = -3.4e38f;
    for (int i = tid; i < Nn; i += blockDim.x) m = fmaxf(m, col[(size_t)i*Nn]);
    red[tid] = m; __syncthreads();
    for (int s = blockDim.x/2; s > 0; s >>= 1) { if (tid < s) red[tid] = fmaxf(red[tid], red[tid+s]); __syncthreads(); }
    m = red[0]; __syncthreads();
    float sum = 0.f;
    for (int i = tid; i < Nn; i += blockDim.x) {
        float e = __expf(col[(size_t)i*Nn] - m);
        col[(size_t)i*Nn] = e; sum += e;
    }
    red[tid] = sum; __syncthreads();
    for (int s = blockDim.x/2; s > 0; s >>= 1) { if (tid < s) red[tid] += red[tid+s]; __syncthreads(); }
    float inv = 1.f / red[0];
    for (int i = tid; i < Nn; i += blockDim.x) col[(size_t)i*Nn] *= inv;
}

// ---------------- Chebyshev graph conv ----------------
__global__ void tax0_kernel(const float* __restrict__ S, const float* __restrict__ X,
                            float* __restrict__ tax0, int t, int f) {
    int idx = GIDX; if (idx >= Bb * Nn * f) return;
    int c = idx % f; int n = (idx / f) % Nn; int b = idx / (f*Nn);
    float diag = S[(size_t)b*Nn*Nn + (size_t)n*Nn + n];
    tax0[idx] = diag * X[((b*Nn + n)*f + c)*Tt + t];
}

__global__ void prop_kernel(const float* __restrict__ S, const float* __restrict__ src,
                            float* __restrict__ dst, const int* __restrict__ edges_t,
                            const float* __restrict__ dinv, int t, int f) {
    int idx = GIDX; if (idx >= Bb * Ee) return;
    int e = idx % Ee, b = idx / Ee;
    int row = edges_t[(t*2 + 0)*Ee + e];
    int col = edges_t[(t*2 + 1)*Ee + e];
    float att = S[(size_t)b*Nn*Nn + (size_t)row*Nn + col] * (-dinv[t*Nn + row] * dinv[t*Nn + col]);
    const float* sp = src + ((size_t)b*Nn + row) * f;
    float* dp = dst + ((size_t)b*Nn + col) * f;
    for (int c = 0; c < f; ++c) atomicAdd(&dp[c], att * sp[c]);
}

__global__ void tax2_fin_kernel(float* __restrict__ tax2, const float* __restrict__ tax0, int f) {
    int idx = GIDX; if (idx >= Bb * Nn * f) return;
    tax2[idx] = 2.f * tax2[idx] - tax0[idx];
}

__global__ void cheb_combine_kernel(const float* __restrict__ tax0, const float* __restrict__ tax1,
                                    const float* __restrict__ tax2, const float* __restrict__ cw,
                                    const float* __restrict__ cb, float* __restrict__ sg,
                                    int t, int f) {
    int idx = GIDX; if (idx >= Bb * Nn * Cc) return;
    int co = idx % Cc; int n = (idx / Cc) % Nn; int b = idx / (Cc*Nn);
    float s = cb[co];
    const float* x0 = tax0 + ((size_t)b*Nn + n) * f;
    const float* x1 = tax1 + ((size_t)b*Nn + n) * f;
    const float* x2 = tax2 + ((size_t)b*Nn + n) * f;
    for (int c = 0; c < f; ++c) {
        s += x0[c] * cw[(0*f + c)*Cc + co];
        s += x1[c] * cw[(1*f + c)*Cc + co];
        s += x2[c] * cw[(2*f + c)*Cc + co];
    }
    sg[((b*Nn + n)*Cc + co)*Tt + t] = fmaxf(s, 0.f);
}

// time conv + residual + relu + layernorm over C; one 64-thread block per (b,n,t)
__global__ void conv_ln_kernel(const float* __restrict__ sg, const float* __restrict__ X,
                               const float* __restrict__ tw, const float* __restrict__ tb,
                               const float* __restrict__ rw, const float* __restrict__ rb,
                               const float* __restrict__ lg, const float* __restrict__ lb,
                               float* __restrict__ out, int f) {
    __shared__ float red[Cc];
    int t = blockIdx.x % Tt; int n = (blockIdx.x / Tt) % Nn; int b = blockIdx.x / (Tt*Nn);
    int c = threadIdx.x;
    float h = tb[c];
    for (int ci = 0; ci < Cc; ++ci) {
        const float* sgp = sg + ((size_t)(b*Nn + n)*Cc + ci) * Tt;
        const float* w3 = tw + (c*Cc + ci) * 3;
        if (t - 1 >= 0) h += sgp[t-1] * w3[0];
        h += sgp[t] * w3[1];
        if (t + 1 < Tt) h += sgp[t+1] * w3[2];
    }
    float r = rb[c];
    for (int ci = 0; ci < f; ++ci) r += X[((b*Nn + n)*f + ci)*Tt + t] * rw[c*f + ci];
    float o = fmaxf(r + h, 0.f);
    red[c] = o; __syncthreads();
    for (int s2 = Cc/2; s2 > 0; s2 >>= 1) { if (c < s2) red[c] += red[c+s2]; __syncthreads(); }
    float mu = red[0] / Cc; __syncthreads();
    float d = o - mu;
    red[c] = d * d; __syncthreads();
    for (int s2 = Cc/2; s2 > 0; s2 >>= 1) { if (c < s2) red[c] += red[c+s2]; __syncthreads(); }
    float var = red[0] / Cc;
    out[((b*Nn + n)*Cc + c)*Tt + t] = d * rsqrtf(var + 1e-5f) * lg[c] + lb[c];
}

// ---------------- head ----------------
__global__ void g_kernel(const float* __restrict__ h, float* __restrict__ g) {
    int idx = GIDX; if (idx >= Bb * Nn) return;
    int n = idx % Nn, b = idx / Nn;
    float s = 0.f;
    for (int c = 0; c < Cc; ++c) s += h[((b*Nn + n)*Cc + c)*Tt + 0];
    g[idx] = s / Cc;
}

__global__ void z_kernel(const float* __restrict__ g, const float* __restrict__ imu,
                         float* __restrict__ z) {
    int idx = GIDX; if (idx >= Bb * DH) return;
    int k = idx % DH, b = idx / DH;
    z[idx] = (k < Nn) ? g[b*Nn + k] : imu[b*24 + (k - Nn)];
}

__global__ void mlp_kernel(const float* __restrict__ in, const float* __restrict__ w,
                           const float* __restrict__ bias, float* __restrict__ out,
                           int K, int M, int doRelu) {
    int idx = GIDX; if (idx >= Bb * M) return;
    int m = idx % M, b = idx / M;
    float s = bias[m];
    for (int k = 0; k < K; ++k) s += in[b*K + k] * w[k*M + m];
    out[idx] = doRelu ? fmaxf(s, 0.f) : s;
}

// ---------------- host ----------------
struct BlockP {
    const float *U1,*U2,*U3,*Ve,*Vs,*W1,*W2,*W3,*be,*bs,
                *cheb_b,*cheb_w,*ln_b,*ln_g,*res_b,*res_w,*time_b,*time_w;
};

static inline dim3 g1(long long n, int tpb = 256) { return dim3((unsigned)((n + tpb - 1) / tpb)); }

extern "C" void kernel_launch(void* const* d_in, const int* in_sizes, int n_in,
                              void* d_out, int out_size, void* d_ws, size_t ws_size,
                              hipStream_t stream) {
    (void)in_sizes; (void)n_in; (void)out_size; (void)ws_size;
    // jax pytree (sorted-key) flattening of setup_inputs() dict
    const int*   edge_index = (const int*)  d_in[0];
    const float* imu        = (const float*)d_in[1];
    BlockP bp[3];
    for (int i = 0; i < 3; ++i) {
        int o = 2 + 18 * i;
        bp[i].U1     = (const float*)d_in[o+0];  bp[i].U2     = (const float*)d_in[o+1];
        bp[i].U3     = (const float*)d_in[o+2];  bp[i].Ve     = (const float*)d_in[o+3];
        bp[i].Vs     = (const float*)d_in[o+4];  bp[i].W1     = (const float*)d_in[o+5];
        bp[i].W2     = (const float*)d_in[o+6];  bp[i].W3     = (const float*)d_in[o+7];
        bp[i].be     = (const float*)d_in[o+8];  bp[i].bs     = (const float*)d_in[o+9];
        bp[i].cheb_b = (const float*)d_in[o+10]; bp[i].cheb_w = (const float*)d_in[o+11];
        bp[i].ln_b   = (const float*)d_in[o+12]; bp[i].ln_g   = (const float*)d_in[o+13];
        bp[i].res_b  = (const float*)d_in[o+14]; bp[i].res_w  = (const float*)d_in[o+15];
        bp[i].time_b = (const float*)d_in[o+16]; bp[i].time_w = (const float*)d_in[o+17];
    }
    const float* head_b1 = (const float*)d_in[56];
    const float* head_b2 = (const float*)d_in[57];
    const float* head_b3 = (const float*)d_in[58];
    const float* head_w1 = (const float*)d_in[59];
    const float* head_w2 = (const float*)d_in[60];
    const float* head_w3 = (const float*)d_in[61];
    const float* pool_w  = (const float*)d_in[62];
    const float* x       = (const float*)d_in[63];

    // workspace carve
    char* wsp = (char*)d_ws;
    auto alloc = [&](size_t bytes) -> void* {
        void* p = (void*)wsp; wsp += (bytes + 255) & ~(size_t)255; return p;
    };
    const size_t actB = (size_t)Bb*Nn*Cc*Tt*sizeof(float);       // 16.8 MB
    float* bufX   = (float*)alloc(actB);
    float* bufY   = (float*)alloc(actB);
    float* xtat   = (float*)alloc(actB);
    float* sg     = (float*)alloc(actB);
    float* tax0   = (float*)alloc((size_t)Bb*Nn*Cc*sizeof(float));
    float* tax1   = (float*)alloc((size_t)Bb*Nn*Cc*sizeof(float));
    float* tax2   = (float*)alloc((size_t)Bb*Nn*Cc*sizeof(float));
    float* Sbuf   = (float*)alloc((size_t)Bb*Nn*Nn*sizeof(float));  // 64 MB
    bf16*  Pt     = (bf16*) alloc((size_t)Bb*Nn*Nn*sizeof(bf16));   // 32 MB
    bf16*  VsBf   = (bf16*) alloc((size_t)Nn*Nn*sizeof(bf16));
    float* scores = (float*)alloc((size_t)Bb*Nn*Tt*sizeof(float));
    float* spbuf  = (float*)alloc((size_t)Bb*Nn*Tt*sizeof(float));
    int*   perm   = (int*)  alloc((size_t)Bb*Nn*Tt*sizeof(int));
    int*   edgesT = (int*)  alloc((size_t)Tt*2*Ee*sizeof(int));
    int*   invp   = (int*)  alloc((size_t)Tt*Nn*sizeof(int));
    float* dinv   = (float*)alloc((size_t)Tt*Nn*sizeof(float));
    float* lhs_tf = (float*)alloc((size_t)Bb*Tt*Cc*sizeof(float));
    float* lhsN   = (float*)alloc((size_t)Bb*Tt*Nn*sizeof(float));
    float* rhsNT  = (float*)alloc((size_t)Bb*Nn*Tt*sizeof(float));
    float* Etb    = (float*)alloc((size_t)Bb*Tt*Tt*sizeof(float));
    float* lhs_s  = (float*)alloc((size_t)Bb*Nn*Tt*sizeof(float));
    float* rhs_s  = (float*)alloc((size_t)Bb*Tt*Nn*sizeof(float));
    float* gbuf   = (float*)alloc((size_t)Bb*Nn*sizeof(float));
    float* zbuf   = (float*)alloc((size_t)Bb*DH*sizeof(float));
    float* z1     = (float*)alloc((size_t)Bb*256*sizeof(float));
    float* z2     = (float*)alloc((size_t)Bb*128*sizeof(float));

    // -------- pooling / permutation --------
    scores_kernel<<<g1((long long)Bb*Nn*Tt), 256, 0, stream>>>(x, pool_w, scores);
    pool_sort_kernel<<<dim3(Bb*Tt), dim3(Nn), 0, stream>>>(scores, spbuf, perm);
    gather_kernel<<<g1((long long)Bb*Nn*Fin*Tt), 256, 0, stream>>>(x, perm, spbuf, bufX);
    inv_kernel<<<g1((long long)Tt*Nn), 256, 0, stream>>>(perm, invp);
    edges_map_kernel<<<g1((long long)Tt*2*Ee), 256, 0, stream>>>(edge_index, invp, edgesT);
    zero_kernel<<<g1((long long)Tt*Nn), 256, 0, stream>>>(dinv, (long long)Tt*Nn);
    deg_kernel<<<g1((long long)Tt*Ee), 256, 0, stream>>>(edgesT, dinv);
    dinv_kernel<<<g1((long long)Tt*Nn), 256, 0, stream>>>(dinv);

    float* cur = bufX; float* nxt = bufY;
    for (int blk = 0; blk < 3; ++blk) {
        const BlockP& p = bp[blk];
        const int f = (blk == 0) ? Fin : Cc;
        // temporal attention
        t_lhs1_kernel<<<g1((long long)Bb*Tt*f), 256, 0, stream>>>(cur, p.U1, lhs_tf, f);
        t_lhs2_kernel<<<g1((long long)Bb*Tt*Nn), 256, 0, stream>>>(lhs_tf, p.U2, lhsN, f);
        t_rhs_kernel<<<g1((long long)Bb*Nn*Tt), 256, 0, stream>>>(cur, p.U3, rhsNT, f);
        et_kernel<<<dim3(Bb), dim3(32), 0, stream>>>(lhsN, rhsNT, p.Ve, p.be, Etb);
        xtat_kernel<<<g1((long long)Bb*Nn*f*Tt), 256, 0, stream>>>(cur, Etb, xtat, f);
        // spatial attention inputs
        lhs_s_kernel<<<g1((long long)Bb*Nn*Tt), 256, 0, stream>>>(xtat, p.W1, p.W2, lhs_s, f);
        rhs_s_kernel<<<g1((long long)Bb*Tt*Nn), 256, 0, stream>>>(xtat, p.W3, rhs_s, f);
        pt_kernel<<<g1((long long)Bb*Nn*Nn), 256, 0, stream>>>(lhs_s, rhs_s, p.bs, Pt);
        f32_to_bf16_kernel<<<g1((long long)Nn*Nn), 256, 0, stream>>>(p.Vs, VsBf, (long long)Nn*Nn);
        // dominant GEMM: S = Vs @ P  (bf16 WMMA, f32 accumulate, LDS-staged B panel)
        wmma_vs_p_kernel<<<dim3(Nn/16, Nn/128, Bb), dim3(256), 0, stream>>>(VsBf, Pt, Sbuf);
        colsoftmax_kernel<<<dim3(Bb*Nn), dim3(128), 0, stream>>>(Sbuf);
        // Chebyshev conv per time step
        for (int t = 0; t < Tt; ++t) {
            tax0_kernel<<<g1((long long)Bb*Nn*f), 256, 0, stream>>>(Sbuf, cur, tax0, t, f);
            zero_kernel<<<g1((long long)Bb*Nn*f), 256, 0, stream>>>(tax1, (long long)Bb*Nn*f);
            prop_kernel<<<g1((long long)Bb*Ee), 256, 0, stream>>>(Sbuf, tax0, tax1, edgesT, dinv, t, f);
            zero_kernel<<<g1((long long)Bb*Nn*f), 256, 0, stream>>>(tax2, (long long)Bb*Nn*f);
            prop_kernel<<<g1((long long)Bb*Ee), 256, 0, stream>>>(Sbuf, tax1, tax2, edgesT, dinv, t, f);
            tax2_fin_kernel<<<g1((long long)Bb*Nn*f), 256, 0, stream>>>(tax2, tax0, f);
            cheb_combine_kernel<<<g1((long long)Bb*Nn*Cc), 256, 0, stream>>>(
                tax0, tax1, tax2, p.cheb_w, p.cheb_b, sg, t, f);
        }
        // time conv + residual + LN
        conv_ln_kernel<<<dim3(Bb*Nn*Tt), dim3(Cc), 0, stream>>>(
            sg, cur, p.time_w, p.time_b, p.res_w, p.res_b, p.ln_g, p.ln_b, nxt, f);
        float* tmp = cur; cur = nxt; nxt = tmp;
    }
    // head
    g_kernel<<<g1((long long)Bb*Nn), 256, 0, stream>>>(cur, gbuf);
    z_kernel<<<g1((long long)Bb*DH), 256, 0, stream>>>(gbuf, imu, zbuf);
    mlp_kernel<<<g1((long long)Bb*256), 256, 0, stream>>>(zbuf, head_w1, head_b1, z1, DH, 256, 1);
    mlp_kernel<<<g1((long long)Bb*128), 256, 0, stream>>>(z1, head_w2, head_b2, z2, 256, 128, 1);
    mlp_kernel<<<g1((long long)Bb*6),   256, 0, stream>>>(z2, head_w3, head_b3, (float*)d_out, 128, 6, 0);
}
